// Decoder_79663053406429
// MI455X (gfx1250) — matrix operand order repaired and checked by reading, not compiled, not run on hardware
//
#include <hip/hip_runtime.h>
#include <hip/hip_bf16.h>
#include <math.h>
#include <stdint.h>

// Problem dims (compile-time constants from the reference)
#define B_   4096
#define S_   24
#define T_   32
#define V_   120
#define E_   128
#define HE_  512
#define HD_  512
#define NG_  2048   // 4*HD
#define KG_  640    // E + HD
#define VP_  128    // vocab padded to multiple of 16 for WMMA

// LDS B-panel stage: 64 output cols x 64 K (bf16) = 8KB data per stage.
// Row = 128 data bytes padded to 144 so 32B lane reads land on distinct
// bank quads (stride 36 dwords mod 64 banks).
#define LDS_ROW_B 144
#define LDS_BUF_B (64 * LDS_ROW_B)

typedef __attribute__((ext_vector_type(16))) __bf16 v16bf;
typedef __attribute__((ext_vector_type(8)))  float  v8f;

union Frag { v16bf v; uint4 q[2]; unsigned int u[8]; };

__device__ inline unsigned short f2bf(float x) {
  unsigned int u = __float_as_uint(x);
  u += 0x7FFFu + ((u >> 16) & 1u);          // round-to-nearest-even
  return (unsigned short)(u >> 16);
}
__device__ inline float bf2f(unsigned short h) {
  return __uint_as_float(((unsigned int)h) << 16);
}
__device__ inline unsigned int pk2(float a, float b) {
  return (unsigned int)f2bf(a) | ((unsigned int)f2bf(b) << 16);
}
__device__ inline float sigm(float x) { return 1.f / (1.f + __expf(-x)); }

// Raw workgroup split-barrier: avoids the compiler's conservative
// s_wait_loadcnt_dscnt 0x0 drain that __syncthreads() inserts, so the
// A-fragment prefetch stays in flight across the barrier. Producer/consumer
// LDS ordering is handled explicitly with s_wait_asynccnt (below).
__device__ inline void wg_barrier() {
  asm volatile("s_barrier_signal -1\n\ts_barrier_wait -1" ::: "memory");
}

// Async-copy the 64x64-bf16 B panel into LDS: 512 x 16B chunks, 2 per thread.
// Tracked by ASYNCcnt (CDNA5 async global->LDS path).
__device__ inline void stage_async64(const unsigned short* BT, int K, int n0,
                                     int kt, unsigned lds_buf, int tid) {
  int n_loc = tid >> 2;            // 0..63 panel row (output column)
  int c0 = tid & 3;                // chunks c0 and c0+4 of 8 x 16B per row
  const unsigned short* g0 = BT + (size_t)(n0 + n_loc) * K + kt + c0 * 8;
  unsigned l0 = lds_buf + (unsigned)(n_loc * LDS_ROW_B + c0 * 16);
  asm volatile("global_load_async_to_lds_b128 %0, %1, off"
               :: "v"(l0), "v"((const void*)g0) : "memory");
  const unsigned short* g1 = g0 + 32;          // +4 chunks
  unsigned l1 = l0 + 64;
  asm volatile("global_load_async_to_lds_b128 %0, %1, off"
               :: "v"(l1), "v"((const void*)g1) : "memory");
}
__device__ inline void wait_async_2() {        // next stage (2 copies) inflight
  asm volatile("s_wait_asynccnt 0x2" ::: "memory");
}
__device__ inline void wait_async_0() {
  asm volatile("s_wait_asynccnt 0x0" ::: "memory");
}

// A-fragment load (ISA 7.12.2 16-bit A layout: two 16B chunks at K+0 / K+16)
template<bool AF32>
__device__ inline void loadA(Frag& a, const void* A, size_t row, int lda,
                             int kt, int lh) {
  if (AF32) {
    const float* ap = (const float*)A + row * lda + kt + lh * 8;
    float4 f0 = *(const float4*)(ap);
    float4 f1 = *(const float4*)(ap + 4);
    float4 f2 = *(const float4*)(ap + 16);
    float4 f3 = *(const float4*)(ap + 20);
    a.u[0] = pk2(f0.x, f0.y); a.u[1] = pk2(f0.z, f0.w);
    a.u[2] = pk2(f1.x, f1.y); a.u[3] = pk2(f1.z, f1.w);
    a.u[4] = pk2(f2.x, f2.y); a.u[5] = pk2(f2.z, f2.w);
    a.u[6] = pk2(f3.x, f3.y); a.u[7] = pk2(f3.z, f3.w);
  } else {
    const unsigned short* ap = (const unsigned short*)A + row * lda + kt + lh * 8;
    a.q[0] = *(const uint4*)(ap);
    a.q[1] = *(const uint4*)(ap + 16);
  }
}

// Read one 16x16x32 B fragment from the staged LDS panel.
__device__ inline void loadB_lds(Frag& b, const char* bbase, int i, int j,
                                 int lane, int lh) {
  const unsigned short* bp = (const unsigned short*)
      (bbase + (i * 16 + (lane & 15)) * LDS_ROW_B + j * 64 + lh * 16);
  b.q[0] = *(const uint4*)(bp);
  b.q[1] = *(const uint4*)(bp + 16);
}

// ---------------------------------------------------------------------------
// bf16 WMMA GEMM:  C(M,N) = act(A(M,K) * B(K,N) + bias)
// BT: N rows of K bf16 (weights pre-transposed). Wave computes a 16x64 strip;
// the workgroup (8 waves) shares a 64(N)x64(K) B panel staged in LDS with
// global_load_async_to_lds_b128, double-buffered: 8 WMMAs per barrier pair.
// Grid: x = N/64, y = M/128; block = 256 threads. K % 64 == 0.
// ---------------------------------------------------------------------------
template<bool AF32, bool BIAS, bool ACT_TANH, bool WF32, bool WBF16>
__global__ __launch_bounds__(256) void k_gemm(
    const void* __restrict__ A, const unsigned short* __restrict__ BT,
    const float* __restrict__ bias, float* __restrict__ Cf,
    unsigned short* __restrict__ Cb, int K, int lda, int ldc)
{
  __shared__ __align__(16) char Bst[2][LDS_BUF_B];
  const int tid  = threadIdx.x;
  const int lane = tid & 31;
  const int wave = tid >> 5;
  const int m0 = blockIdx.y * 128 + wave * 16;
  const int n0 = blockIdx.x * 64;
  const size_t row = (size_t)(m0 + (lane & 15));
  const int lh  = lane >> 4;                 // 0: lanes 0-15, 1: lanes 16-31
  const unsigned lds0 = (unsigned)(uintptr_t)&Bst[0][0];
  const unsigned lds1 = (unsigned)(uintptr_t)&Bst[1][0];
  v8f acc[4] = {};

  stage_async64(BT, K, n0, 0, lds0, tid);    // prologue: stage k=0..63
  Frag a_cur[2];
  loadA<AF32>(a_cur[0], A, row, lda, 0, lh);
  loadA<AF32>(a_cur[1], A, row, lda, 32, lh);

  for (int kt = 0; kt < K; kt += 64) {
    const int par = (kt >> 6) & 1;
    if (kt + 64 < K) {                       // issue next stage, keep 2 inflight
      stage_async64(BT, K, n0, kt + 64, par ? lds0 : lds1, tid);
      wait_async_2();
    } else {
      wait_async_0();
    }
    wg_barrier();                            // panel[par] visible to all waves
    Frag a_nxt[2];
    if (kt + 64 < K) {
      loadA<AF32>(a_nxt[0], A, row, lda, kt + 64, lh);
      loadA<AF32>(a_nxt[1], A, row, lda, kt + 96, lh);
    }
    const char* bbase = Bst[par];
#pragma unroll
    for (int j = 0; j < 2; j++) {
#pragma unroll
      for (int i = 0; i < 4; i++) {
        Frag b;
        loadB_lds(b, bbase, i, j, lane, lh);
        acc[i] = __builtin_amdgcn_wmma_f32_16x16x32_bf16(
            false, a_cur[j].v, false, b.v, (short)0, acc[i], false, false);
      }
    }
    wg_barrier();                            // readers done before panel reuse
    if (kt + 64 < K) { a_cur[0] = a_nxt[0]; a_cur[1] = a_nxt[1]; }
  }
#pragma unroll
  for (int i = 0; i < 4; i++) {
    int col = n0 + i * 16 + (lane & 15);
    float bv = BIAS ? bias[col] : 0.f;
#pragma unroll
    for (int r = 0; r < 8; r++) {
      float val = acc[i][r] + bv;
      if (ACT_TANH) val = tanhf(val);
      size_t o = (size_t)(m0 + lh * 8 + r) * ldc + col;
      if (WF32)  Cf[o] = val;
      if (WBF16) Cb[o] = f2bf(val);
    }
  }
}

// Gates GEMM: A = [emb_t (B x 128) | h (B x 512)] split at K=128 (a multiple
// of the 64-wide stage, so a stage never straddles the two sources).
__global__ __launch_bounds__(256) void k_gates(
    const unsigned short* __restrict__ X, const unsigned short* __restrict__ H,
    const unsigned short* __restrict__ BT, const float* __restrict__ bias,
    float* __restrict__ Cf)
{
  __shared__ __align__(16) char Bst[2][LDS_BUF_B];
  const int tid  = threadIdx.x;
  const int lane = tid & 31;
  const int wave = tid >> 5;
  const int m0 = blockIdx.y * 128 + wave * 16;
  const int n0 = blockIdx.x * 64;
  const size_t row = (size_t)(m0 + (lane & 15));
  const int lh  = lane >> 4;
  const unsigned lds0 = (unsigned)(uintptr_t)&Bst[0][0];
  const unsigned lds1 = (unsigned)(uintptr_t)&Bst[1][0];
  v8f acc[4] = {};

  stage_async64(BT, KG_, n0, 0, lds0, tid);
  Frag a_cur[2];
  loadA<false>(a_cur[0], X, row, E_, 0, lh);
  loadA<false>(a_cur[1], X, row, E_, 32, lh);

  for (int kk = 0; kk < KG_; kk += 64) {
    const int par = (kk >> 6) & 1;
    if (kk + 64 < KG_) {
      stage_async64(BT, KG_, n0, kk + 64, par ? lds0 : lds1, tid);
      wait_async_2();
    } else {
      wait_async_0();
    }
    wg_barrier();
    Frag a_nxt[2];
    if (kk + 64 < KG_) {
      int kn = kk + 64;
      if (kn < E_) {
        loadA<false>(a_nxt[0], X, row, E_, kn, lh);
        loadA<false>(a_nxt[1], X, row, E_, kn + 32, lh);
      } else {
        loadA<false>(a_nxt[0], H, row, HD_, kn - E_, lh);
        loadA<false>(a_nxt[1], H, row, HD_, kn - E_ + 32, lh);
      }
    }
    const char* bbase = Bst[par];
#pragma unroll
    for (int j = 0; j < 2; j++) {
#pragma unroll
      for (int i = 0; i < 4; i++) {
        Frag b;
        loadB_lds(b, bbase, i, j, lane, lh);
        acc[i] = __builtin_amdgcn_wmma_f32_16x16x32_bf16(
            false, a_cur[j].v, false, b.v, (short)0, acc[i], false, false);
      }
    }
    wg_barrier();
    if (kk + 64 < KG_) { a_cur[0] = a_nxt[0]; a_cur[1] = a_nxt[1]; }
  }
#pragma unroll
  for (int i = 0; i < 4; i++) {
    int col = n0 + i * 16 + (lane & 15);
    float bv = bias[col];
#pragma unroll
    for (int r = 0; r < 8; r++)
      Cf[(size_t)(m0 + lh * 8 + r) * NG_ + col] = acc[i][r] + bv;
  }
}

// LSTM pointwise: torch gate order i,f,g,o laid out in N chunks of 512.
__global__ void k_lstm(const float* __restrict__ gates, float* __restrict__ c,
                       unsigned short* __restrict__ hb,
                       unsigned short* __restrict__ aap)
{
  int id = blockIdx.x * 256 + threadIdx.x;
  if (id >= B_ * HD_) return;
  int b = id / HD_, j = id - b * HD_;
  const float* gr = gates + (size_t)b * NG_;
  float gi = gr[j], gf = gr[HD_ + j], gg = gr[2 * HD_ + j], go = gr[3 * HD_ + j];
  float cn = sigm(gf) * c[id] + sigm(gi) * tanhf(gg);
  float hn = sigm(go) * tanhf(cn);
  c[id] = cn;
  unsigned short hv = f2bf(hn);
  hb[id] = hv;
  aap[(size_t)b * (HD_ + HE_) + j] = hv;  // [h | ctx] concat buffer, h half
}

// Bahdanau score + softmax + context. One block per batch row (256 thr).
__global__ __launch_bounds__(256) void k_attn(
    const float* __restrict__ q, const unsigned short* __restrict__ encproj,
    const float* __restrict__ enc, const float* __restrict__ v_attn,
    float* __restrict__ attn_buf, float* __restrict__ out_attn, int t,
    float* __restrict__ ctx, unsigned short* __restrict__ aap)
{
  __shared__ float qs[HD_];
  __shared__ float sc[S_];
  int b = blockIdx.x;
  int lane = threadIdx.x & 31;
  int wave = threadIdx.x >> 5;
  for (int i = threadIdx.x; i < HD_; i += 256) qs[i] = q[(size_t)b * HD_ + i];
  __syncthreads();
  // 8 waves x 3 scores = 24; lane-parallel dot over HD, shuffle reduce (wave32)
  for (int si = 0; si < 3; si++) {
    int s = wave * 3 + si;
    const unsigned short* ep = encproj + ((size_t)b * S_ + s) * HD_;
    float acc = 0.f;
    for (int k = lane; k < HD_; k += 32)
      acc += tanhf(bf2f(ep[k]) + qs[k]) * v_attn[k];
    for (int o = 16; o > 0; o >>= 1) acc += __shfl_xor(acc, o, 32);
    if (lane == 0) sc[s] = acc;
  }
  __syncthreads();
  if (threadIdx.x < 32) {
    float v = (lane < S_) ? sc[lane] : -3.0e38f;
    float m = v;
    for (int o = 16; o > 0; o >>= 1) m = fmaxf(m, __shfl_xor(m, o, 32));
    float e = (lane < S_) ? __expf(v - m) : 0.f;
    float ss = e;
    for (int o = 16; o > 0; o >>= 1) ss += __shfl_xor(ss, o, 32);
    float a = e / ss;
    if (lane < S_) {
      sc[lane] = a;
      attn_buf[(size_t)b * S_ + lane] = a;
      out_attn[((size_t)b * T_ + t) * S_ + lane] = a;
    }
  }
  __syncthreads();
  // ctx = attn @ enc   (per-batch matvec; enc hot in L2)
  for (int e = threadIdx.x; e < HE_; e += 256) {
    const float* eb = enc + (size_t)b * S_ * HE_ + e;
    float s = 0.f;
#pragma unroll
    for (int si = 0; si < S_; si++) s += sc[si] * eb[(size_t)si * HE_];
    ctx[(size_t)b * HE_ + e] = s;
    aap[(size_t)b * (HD_ + HE_) + HD_ + e] = f2bf(s);
  }
}

// Fused vocab softmax + pointer-generator blend. One wave per batch row.
__global__ __launch_bounds__(256) void k_out(
    const float* __restrict__ logits, const float* __restrict__ ah,
    const float* __restrict__ ctx, const float* __restrict__ attn_buf,
    const int* __restrict__ in_ids, const float* __restrict__ w_copy,
    const float* __restrict__ b_copy, float* __restrict__ outv, int t)
{
  int gw = (blockIdx.x * 256 + threadIdx.x) >> 5;
  int lane = threadIdx.x & 31;
  if (gw >= B_) return;
  float d = 0.f;
  for (int k = lane; k < HD_; k += 32) d += ah[(size_t)gw * HD_ + k] * w_copy[k];
  for (int k = lane; k < HE_; k += 32) d += ctx[(size_t)gw * HE_ + k] * w_copy[HD_ + k];
  for (int o = 16; o > 0; o >>= 1) d += __shfl_xor(d, o, 32);
  float p = sigm(d + b_copy[0]);
  float lg[4], ex[4];
  float m = -3.0e38f;
#pragma unroll
  for (int c2 = 0; c2 < 4; c2++) {
    int v = lane + 32 * c2;
    lg[c2] = (v < V_) ? logits[(size_t)gw * VP_ + v] : -3.0e38f;
    m = fmaxf(m, lg[c2]);
  }
  for (int o = 16; o > 0; o >>= 1) m = fmaxf(m, __shfl_xor(m, o, 32));
  float ss = 0.f;
#pragma unroll
  for (int c2 = 0; c2 < 4; c2++) {
    int v = lane + 32 * c2;
    ex[c2] = (v < V_) ? __expf(lg[c2] - m) : 0.f;
    ss += ex[c2];
  }
  for (int o = 16; o > 0; o >>= 1) ss += __shfl_xor(ss, o, 32);
  float cd[4] = {0.f, 0.f, 0.f, 0.f};
  for (int s = 0; s < S_; s++) {
    int id = in_ids[(size_t)gw * S_ + s];
    float a = attn_buf[(size_t)gw * S_ + s];
#pragma unroll
    for (int c2 = 0; c2 < 4; c2++)
      if (id == lane + 32 * c2) cd[c2] += a;
  }
#pragma unroll
  for (int c2 = 0; c2 < 4; c2++) {
    int v = lane + 32 * c2;
    if (v < V_)
      outv[((size_t)gw * T_ + t) * V_ + v] = (1.f - p) * (ex[c2] / ss) + p * cd[c2];
  }
}

// ------------------------ one-time packing kernels -------------------------
__global__ void k_pack_wgT(unsigned short* dst, const float* W_ih, const float* W_hh) {
  int id = blockIdx.x * 256 + threadIdx.x;
  if (id >= NG_ * KG_) return;
  int n = id / KG_, k = id - n * KG_;
  float v = (k < E_) ? W_ih[(size_t)n * E_ + k] : W_hh[(size_t)n * HD_ + (k - E_)];
  dst[id] = f2bf(v);
}
__global__ void k_transpose512(unsigned short* dst, const float* src) {
  int id = blockIdx.x * 256 + threadIdx.x;
  if (id >= 512 * 512) return;
  int n = id >> 9, k = id & 511;
  dst[id] = f2bf(src[(size_t)k * 512 + n]);
}
__global__ void k_pack_direct(unsigned short* dst, const float* src, int n) {
  int id = blockIdx.x * 256 + threadIdx.x;
  if (id < n) dst[id] = f2bf(src[id]);
}
__global__ void k_pack_wout(unsigned short* dst, const float* W_out) {
  int id = blockIdx.x * 256 + threadIdx.x;
  if (id >= VP_ * HD_) return;
  int r = id / HD_, c2 = id - r * HD_;
  dst[id] = f2bf(r < V_ ? W_out[(size_t)r * HD_ + c2] : 0.f);
}
__global__ void k_pack_bias(float* bg, float* boutp, const float* b_ih,
                            const float* b_hh, const float* b_out) {
  int id = blockIdx.x * 256 + threadIdx.x;
  if (id < NG_) bg[id] = b_ih[id] + b_hh[id];
  else if (id < NG_ + VP_) {
    int j = id - NG_;
    boutp[j] = (j < V_) ? b_out[j] : 0.f;
  }
}
__global__ void k_embed(unsigned short* X, const int* tgt, const float* emb) {
  int id = blockIdx.x * 256 + threadIdx.x;
  if (id >= T_ * B_ * E_) return;
  int t = id / (B_ * E_);
  int r = id - t * (B_ * E_);
  int b = r / E_, e = r - b * E_;
  int tok = (t == 0) ? 2 : tgt[(size_t)b * T_ + (t - 1)];
  X[id] = f2bf(emb[(size_t)tok * E_ + e]);
}
__global__ void k_zero(float* p, int n) {
  int id = blockIdx.x * 256 + threadIdx.x;
  if (id < n) p[id] = 0.f;
}

// ---------------------------------------------------------------------------
extern "C" void kernel_launch(void* const* d_in, const int* in_sizes, int n_in,
                              void* d_out, int out_size, void* d_ws, size_t ws_size,
                              hipStream_t stream) {
  (void)in_sizes; (void)n_in; (void)out_size; (void)ws_size;
  const float* enc     = (const float*)d_in[0];
  const float* enc_fin = (const float*)d_in[1];
  const int*   tgt     = (const int*)d_in[2];
  const int*   in_ids  = (const int*)d_in[3];
  const float* emb     = (const float*)d_in[4];
  const float* W_init  = (const float*)d_in[5];
  const float* b_init  = (const float*)d_in[6];
  const float* W_ih    = (const float*)d_in[7];
  const float* W_hh    = (const float*)d_in[8];
  const float* b_ih    = (const float*)d_in[9];
  const float* b_hh    = (const float*)d_in[10];
  const float* W_enc   = (const float*)d_in[11];
  const float* W_dec   = (const float*)d_in[12];
  const float* b_attn  = (const float*)d_in[13];
  const float* v_attn  = (const float*)d_in[14];
  const float* W_ap    = (const float*)d_in[15];
  const float* b_ap    = (const float*)d_in[16];
  const float* W_out   = (const float*)d_in[17];
  const float* b_out   = (const float*)d_in[18];
  const float* w_copy  = (const float*)d_in[19];
  const float* b_copy  = (const float*)d_in[20];

  float* out_f    = (float*)d_out;
  float* out_attn = out_f + (size_t)B_ * T_ * V_;   // tuple: outputs, attentions

  // workspace carve (~226 MB total)
  char* w = (char*)d_ws;
  size_t off = 0;
  auto carve = [&](size_t bytes) -> char* {
    char* p = w + off;
    off += (bytes + 255) & ~(size_t)255;
    return p;
  };
  unsigned short* WgT     = (unsigned short*)carve((size_t)NG_ * KG_ * 2);
  unsigned short* WencT   = (unsigned short*)carve((size_t)HD_ * HE_ * 2);
  unsigned short* WdecT   = (unsigned short*)carve((size_t)HD_ * HD_ * 2);
  unsigned short* WapB    = (unsigned short*)carve((size_t)HD_ * (HD_ + HE_) * 2);
  unsigned short* WoutB   = (unsigned short*)carve((size_t)VP_ * HD_ * 2);
  unsigned short* WinitB  = (unsigned short*)carve((size_t)HD_ * HE_ * 2);
  float*          bg      = (float*)carve((size_t)NG_ * 4);
  float*          boutp   = (float*)carve((size_t)VP_ * 4);
  unsigned short* Xemb    = (unsigned short*)carve((size_t)T_ * B_ * E_ * 2);
  unsigned short* encproj = (unsigned short*)carve((size_t)B_ * S_ * HD_ * 2);
  float*          gates   = (float*)carve((size_t)B_ * NG_ * 4);
  unsigned short* h_bf    = (unsigned short*)carve((size_t)B_ * HD_ * 2);
  float*          c_st    = (float*)carve((size_t)B_ * HD_ * 4);
  float*          q_buf   = (float*)carve((size_t)B_ * HD_ * 4);
  unsigned short* Aap     = (unsigned short*)carve((size_t)B_ * (HD_ + HE_) * 2);
  float*          ctx     = (float*)carve((size_t)B_ * HE_ * 4);
  float*          ah      = (float*)carve((size_t)B_ * HD_ * 4);
  unsigned short* ahb     = (unsigned short*)carve((size_t)B_ * HD_ * 2);
  float*          attn_b  = (float*)carve((size_t)B_ * S_ * 4);
  float*          logits  = (float*)carve((size_t)B_ * VP_ * 4);

  // ---- one-time prep ----
  k_pack_wgT<<<(NG_ * KG_ + 255) / 256, 256, 0, stream>>>(WgT, W_ih, W_hh);
  k_transpose512<<<(512 * 512 + 255) / 256, 256, 0, stream>>>(WencT, W_enc);
  k_transpose512<<<(512 * 512 + 255) / 256, 256, 0, stream>>>(WdecT, W_dec);
  k_pack_direct<<<(HD_ * (HD_ + HE_) + 255) / 256, 256, 0, stream>>>(WapB, W_ap, HD_ * (HD_ + HE_));
  k_pack_direct<<<(HD_ * HE_ + 255) / 256, 256, 0, stream>>>(WinitB, W_init, HD_ * HE_);
  k_pack_wout<<<(VP_ * HD_ + 255) / 256, 256, 0, stream>>>(WoutB, W_out);
  k_pack_bias<<<(NG_ + VP_ + 255) / 256, 256, 0, stream>>>(bg, boutp, b_ih, b_hh, b_out);
  k_embed<<<(T_ * B_ * E_ + 255) / 256, 256, 0, stream>>>(Xemb, tgt, emb);
  k_zero<<<(B_ * HD_ + 255) / 256, 256, 0, stream>>>(c_st, B_ * HD_);

  // h0 = enc_final @ W_init^T + b_init      (fp32 A -> bf16 h)
  k_gemm<true, true, false, false, true>
      <<<dim3(HD_ / 64, B_ / 128), 256, 0, stream>>>(
          enc_fin, WinitB, b_init, nullptr, h_bf, HE_, HE_, HD_);
  // enc_proj = enc @ W_enc_attn             (bf16, L2-resident 96 MB)
  k_gemm<true, false, false, false, true>
      <<<dim3(HD_ / 64, (B_ * S_) / 128), 256, 0, stream>>>(
          enc, WencT, nullptr, nullptr, encproj, HE_, HE_, HD_);

  // ---- sequential decode, 32 steps ----
  for (int t = 0; t < T_; t++) {
    k_gates<<<dim3(NG_ / 64, B_ / 128), 256, 0, stream>>>(
        Xemb + (size_t)t * B_ * E_, h_bf, WgT, bg, gates);
    k_lstm<<<(B_ * HD_ + 255) / 256, 256, 0, stream>>>(gates, c_st, h_bf, Aap);
    // q = h @ W_dec_attn + b_attn
    k_gemm<false, true, false, true, false>
        <<<dim3(HD_ / 64, B_ / 128), 256, 0, stream>>>(
            h_bf, WdecT, b_attn, q_buf, nullptr, HD_, HD_, HD_);
    k_attn<<<B_, 256, 0, stream>>>(q_buf, encproj, enc, v_attn,
                                   attn_b, out_attn, t, ctx, Aap);
    // ah = tanh([h|ctx] @ W_ap^T + b_ap)
    k_gemm<false, true, true, true, true>
        <<<dim3(HD_ / 64, B_ / 128), 256, 0, stream>>>(
            Aap, WapB, b_ap, ah, ahb, HD_ + HE_, HD_ + HE_, HD_);
    // logits = ah @ W_out^T + b_out (padded N=128)
    k_gemm<false, true, false, true, false>
        <<<dim3(VP_ / 64, B_ / 128), 256, 0, stream>>>(
            ahb, WoutB, boutp, logits, nullptr, HD_, HD_, VP_);
    k_out<<<(B_ * 32 + 255) / 256, 256, 0, stream>>>(
        logits, ah, ctx, attn_b, in_ids, w_copy, b_copy, out_f, t);
  }
}
